// IncMPNEncoder_4252017623666
// MI455X (gfx1250) — matrix operand order repaired and checked by reading, not compiled, not run on hardware
//
#include <hip/hip_runtime.h>
#include <hip/hip_bf16.h>

// ---------------- problem constants (from reference) ----------------
constexpr int E_  = 150000;   // messages
constexpr int S_  = 60000;    // sub-messages (divisible by 16 -> 3750 tiles)
constexpr int NB_ = 6;        // neighbors
constexpr int H_  = 256;      // hidden
constexpr int IN_ = 384;      // fmess dim
constexpr int FD_ = 128;      // node fdim
constexpr int NN_ = 150000;   // num nodes
constexpr int DEPTH_ = 3;

constexpr int MT_ = S_ / 16;  // 3750 M-tiles

// LDS tile strides (padded for bank-conflict-free fragment loads)
constexpr int LDF_H   = H_ + 4;    // f32 tile stride, K=256
constexpr int LDB_H   = H_ + 8;    // bf16 tile stride, K=256 (stride%8==0 -> 16B rows)
constexpr int LDB_384 = IN_ + 8;   // bf16 tile stride, K=384

// ---------------- WMMA types / helpers ----------------
typedef __attribute__((ext_vector_type(16))) __bf16 v16bf;
typedef __attribute__((ext_vector_type(8)))  float  v8f;

__device__ __forceinline__ unsigned short f2bf(float f) {
  union { float f; unsigned u; } c; c.f = f;
  unsigned r = c.u + 0x7FFFu + ((c.u >> 16) & 1u);   // round-to-nearest-even
  return (unsigned short)(r >> 16);
}

__device__ __forceinline__ void store_bf4(unsigned short* dst, float4 v) {
  union { unsigned short s[4]; uint2 u; } t;
  t.s[0] = f2bf(v.x); t.s[1] = f2bf(v.y); t.s[2] = f2bf(v.z); t.s[3] = f2bf(v.w);
  *(uint2*)dst = t.u;   // 8B aligned: col%4==0, stride%8==0
}

// A fragment: 16x32 bf16 (M x K) from a bf16 LDS tile (row-major, stride ld).
// k order used here (same as B): lane L (m=L&15, half=L>>4): elems 0..15 -> K = kb+half*16+0..15
// => one contiguous 32B run (two 16B loads).
__device__ __forceinline__ v16bf load_a_frag_lds(const unsigned short* At, int ld,
                                                 int kb, int lane) {
  int m = lane & 15, half = lane >> 4;
  const unsigned short* p = At + m * ld + kb + half * 16;
  union { v16bf v; uint4 q[2]; } a;
  a.q[0] = *(const uint4*)p;
  a.q[1] = *(const uint4*)(p + 8);
  return a.v;
}

// B fragment: 32x16 bf16 (K x N), element (k,n) = W[n][koff+k], W row-major bf16.
// lane L: n = nbase+(L&15), k = (L>>4)*16 + 0..15 + kb -> 16 contiguous bf16.
__device__ __forceinline__ v16bf load_b_frag(const unsigned short* W, int ldw, int koff,
                                             int kb, int nbase, int lane) {
  int n     = nbase + (lane & 15);
  int khalf = (lane >> 4) * 16;
  const unsigned* p = (const unsigned*)(W + (size_t)n * ldw + koff + kb + khalf);
  union { v16bf v; unsigned u[8]; } b;
#pragma unroll
  for (int d = 0; d < 8; ++d) b.u[d] = p[d];
  return b.v;
}

#define WMMA_BF16(acc, a, b)                                                      \
  acc = __builtin_amdgcn_wmma_f32_16x16x32_bf16(false, (a), false, (b), (short)0, \
                                                (acc), false, false)

__device__ __forceinline__ float sigmoidf_(float x) { return 1.0f / (1.0f + __expf(-x)); }

// ---------------- utility kernels ----------------
__global__ void k_cvt_bf16(const float* __restrict__ src, unsigned short* __restrict__ dst, int n) {
  int i = blockIdx.x * blockDim.x + threadIdx.x;
  if (i < n) dst[i] = f2bf(src[i]);
}

__global__ void k_copy_f4(const float4* __restrict__ src, float4* __restrict__ dst, int n4) {
  int i = blockIdx.x * blockDim.x + threadIdx.x;
  if (i < n4) dst[i] = src[i];
}

__global__ void k_zero_f4(float4* __restrict__ dst, int n4) {
  int i = blockIdx.x * blockDim.x + threadIdx.x;
  if (i < n4) dst[i] = make_float4(0.f, 0.f, 0.f, 0.f);
}

// h[submess] = 0 (initial mask); float4 per thread
__global__ void k_zero_rows(float* __restrict__ hwork, const int* __restrict__ submess, int total4) {
  int i = blockIdx.x * blockDim.x + threadIdx.x;
  if (i < total4) {
    int s = i >> 6;               // / (H_/4)
    int c = (i & 63) * 4;
    *(float4*)&hwork[(size_t)submess[s] * H_ + c] = make_float4(0.f, 0.f, 0.f, 0.f);
  }
}

// ---------------- loop-invariant projections: xz/xr/xh = fmess @ W*^T ----------------
__global__ __launch_bounds__(512)
void k_xproj(const float* __restrict__ fmess,
             const unsigned short* __restrict__ Wz_bf,
             const unsigned short* __restrict__ Wr_bf,
             const unsigned short* __restrict__ Wh_bf,
             float* __restrict__ xz, float* __restrict__ xr, float* __restrict__ xh) {
  __shared__ __align__(16) unsigned short Atb[16 * LDB_384];
  const int tid = threadIdx.x, stile = blockIdx.x;
  for (int i = tid; i < 16 * (IN_ / 4); i += 512) {   // 1536 quads
    int m = i / (IN_ / 4), c = (i % (IN_ / 4)) * 4;
    float4 v = *(const float4*)&fmess[(size_t)(stile * 16 + m) * IN_ + c];
    store_bf4(&Atb[m * LDB_384 + c], v);
  }
  __syncthreads();
  const int lane = tid & 31, wid = tid >> 5;      // wid = N-tile (H_/16 == 16 waves)
  v8f az = {}, ar = {}, ah = {};
#pragma unroll
  for (int kc = 0; kc < IN_ / 32; ++kc) {
    v16bf a  = load_a_frag_lds(Atb, LDB_384, kc * 32, lane);
    v16bf bz = load_b_frag(Wz_bf, IN_ + H_, 0, kc * 32, wid * 16, lane);
    WMMA_BF16(az, a, bz);
    v16bf br = load_b_frag(Wr_bf, IN_, 0, kc * 32, wid * 16, lane);
    WMMA_BF16(ar, a, br);
    v16bf bh = load_b_frag(Wh_bf, IN_ + H_, 0, kc * 32, wid * 16, lane);
    WMMA_BF16(ah, a, bh);
  }
  const int ncol = wid * 16 + (lane & 15);
  const int mb   = (lane >> 4) * 8;
#pragma unroll
  for (int r = 0; r < 8; ++r) {
    size_t row = (size_t)stile * 16 + mb + r;
    xz[row * H_ + ncol] = az[r];
    xr[row * H_ + ncol] = ar[r];
    xh[row * H_ + ncol] = ah[r];
  }
}

// ---------------- fused gather + r-gate GEMM: sum_h, sum_gh ----------------
__global__ __launch_bounds__(512)
void k_msg_pass(const float* __restrict__ hwork, const int* __restrict__ bgraph,
                const float* __restrict__ xr, const float* __restrict__ bu,
                const unsigned short* __restrict__ Ur_bf,
                float* __restrict__ sum_h, float* __restrict__ sum_gh) {
  __shared__ __align__(16) float          Htf[16 * LDF_H];  // gathered h_nei (f32)
  __shared__ __align__(16) unsigned short Htb[16 * LDB_H];  // same tile in bf16
  __shared__ int rowbuf[16];
  const int tid = threadIdx.x, stile = blockIdx.x;
  const int lane = tid & 31, wid = tid >> 5;
  const int ncol = wid * 16 + (lane & 15);
  const int mb   = (lane >> 4) * 8;

  // j-invariant: Ur B-fragments and (xr + bu) epilogue operands
  v16bf B[H_ / 32];
#pragma unroll
  for (int kc = 0; kc < H_ / 32; ++kc)
    B[kc] = load_b_frag(Ur_bf, H_, 0, kc * 32, wid * 16, lane);
  const float bun = bu[ncol];
  v8f xr_reg;
#pragma unroll
  for (int r = 0; r < 8; ++r)
    xr_reg[r] = xr[(size_t)(stile * 16 + mb + r) * H_ + ncol] + bun;

  v8f s_h = {}, s_gh = {};
  for (int j = 0; j < NB_; ++j) {
    if (tid < 16) rowbuf[tid] = bgraph[(stile * 16 + tid) * NB_ + j];
    __syncthreads();   // guards Ht reuse vs previous iteration's epilogue too
    for (int i = tid; i < 16 * (H_ / 4); i += 512) {   // 1024 quads
      int m = i >> 6, c = (i & 63) * 4;
      float4 v = *(const float4*)&hwork[(size_t)rowbuf[m] * H_ + c];
      *(float4*)&Htf[m * LDF_H + c] = v;
      store_bf4(&Htb[m * LDB_H + c], v);
    }
    __syncthreads();
    v8f r2 = {};
#pragma unroll
    for (int kc = 0; kc < H_ / 32; ++kc) {
      v16bf a = load_a_frag_lds(Htb, LDB_H, kc * 32, lane);
      WMMA_BF16(r2, a, B[kc]);
    }
#pragma unroll
    for (int r = 0; r < 8; ++r) {
      float hn = Htf[(mb + r) * LDF_H + ncol];
      float rr = sigmoidf_(r2[r] + xr_reg[r]);
      s_h[r]  += hn;
      s_gh[r] += rr * hn;
    }
  }
#pragma unroll
  for (int r = 0; r < 8; ++r) {
    size_t row = (size_t)stile * 16 + mb + r;
    sum_h[row * H_ + ncol]  = s_h[r];
    sum_gh[row * H_ + ncol] = s_gh[r];
  }
}

// ---------------- fused z-gate / candidate GEMMs + GRU update + scatter ----------------
__global__ __launch_bounds__(512)
void k_gru_update(const float* __restrict__ sum_h, const float* __restrict__ sum_gh,
                  const float* __restrict__ xz, const float* __restrict__ xh,
                  const unsigned short* __restrict__ Wz_bf,
                  const unsigned short* __restrict__ Wh_bf,
                  const float* __restrict__ bz, const float* __restrict__ bh,
                  const int* __restrict__ submess, float* __restrict__ hwork) {
  __shared__ __align__(16) float          SHf[16 * LDF_H];
  __shared__ __align__(16) unsigned short SHb[16 * LDB_H];
  __shared__ __align__(16) unsigned short SGb[16 * LDB_H];
  const int tid = threadIdx.x, stile = blockIdx.x;
  for (int i = tid; i < 16 * (H_ / 4); i += 512) {
    int m = i >> 6, c = (i & 63) * 4;
    size_t idx = (size_t)(stile * 16 + m) * H_ + c;
    float4 a = *(const float4*)&sum_h[idx];
    float4 g = *(const float4*)&sum_gh[idx];
    *(float4*)&SHf[m * LDF_H + c] = a;
    store_bf4(&SHb[m * LDB_H + c], a);
    store_bf4(&SGb[m * LDB_H + c], g);
  }
  __syncthreads();
  const int lane = tid & 31, wid = tid >> 5;
  v8f az = {}, ah = {};
#pragma unroll
  for (int kc = 0; kc < H_ / 32; ++kc) {
    v16bf a1 = load_a_frag_lds(SHb, LDB_H, kc * 32, lane);
    v16bf b1 = load_b_frag(Wz_bf, IN_ + H_, IN_, kc * 32, wid * 16, lane);
    WMMA_BF16(az, a1, b1);
    v16bf a2 = load_a_frag_lds(SGb, LDB_H, kc * 32, lane);
    v16bf b2 = load_b_frag(Wh_bf, IN_ + H_, IN_, kc * 32, wid * 16, lane);
    WMMA_BF16(ah, a2, b2);
  }
  const int ncol = wid * 16 + (lane & 15);
  const int mb   = (lane >> 4) * 8;
  const float bzn = bz[ncol], bhn = bh[ncol];
#pragma unroll
  for (int r = 0; r < 8; ++r) {
    int    m    = mb + r;
    size_t srow = (size_t)stile * 16 + m;
    float  z    = sigmoidf_(az[r] + xz[srow * H_ + ncol] + bzn);
    float  ph   = tanhf(ah[r] + xh[srow * H_ + ncol] + bhn);
    float  sh   = SHf[m * LDF_H + ncol];
    int    drow = submess[srow];
    hwork[(size_t)drow * H_ + ncol] = (1.f - z) * sh + z * ph;
  }
}

// ---------------- readout: relu(concat(fnode, sum_j h[agraph]) @ Wo^T + bo) -> scatter ----------------
__global__ __launch_bounds__(512)
void k_readout(const float* __restrict__ fnode, const float* __restrict__ hwork,
               const int* __restrict__ agraph,
               const unsigned short* __restrict__ Wo_bf, const float* __restrict__ bo,
               const int* __restrict__ subnode, float* __restrict__ node_out) {
  __shared__ __align__(16) unsigned short Atb[16 * LDB_384];
  __shared__ int arow[16 * NB_];
  const int tid = threadIdx.x, stile = blockIdx.x;
  if (tid < 16 * NB_) arow[tid] = agraph[stile * 16 * NB_ + tid];
  // fnode part: cols 0..127
  for (int i = tid; i < 16 * (FD_ / 4); i += 512) {   // 512 quads
    int m = i >> 5, c = (i & 31) * 4;
    float4 v = *(const float4*)&fnode[(size_t)(stile * 16 + m) * FD_ + c];
    store_bf4(&Atb[m * LDB_384 + c], v);
  }
  __syncthreads();
  // neighbor-sum part: cols 128..383
  for (int i = tid; i < 16 * (H_ / 4); i += 512) {    // 1024 quads
    int m = i >> 6, c = (i & 63) * 4;
    float4 acc = make_float4(0.f, 0.f, 0.f, 0.f);
#pragma unroll
    for (int j = 0; j < NB_; ++j) {
      const float4 v = *(const float4*)&hwork[(size_t)arow[m * NB_ + j] * H_ + c];
      acc.x += v.x; acc.y += v.y; acc.z += v.z; acc.w += v.w;
    }
    store_bf4(&Atb[m * LDB_384 + FD_ + c], acc);
  }
  __syncthreads();
  const int lane = tid & 31, wid = tid >> 5;
  v8f ao = {};
#pragma unroll
  for (int kc = 0; kc < (FD_ + H_) / 32; ++kc) {
    v16bf a = load_a_frag_lds(Atb, LDB_384, kc * 32, lane);
    v16bf b = load_b_frag(Wo_bf, FD_ + H_, 0, kc * 32, wid * 16, lane);
    WMMA_BF16(ao, a, b);
  }
  const int ncol = wid * 16 + (lane & 15);
  const int mb   = (lane >> 4) * 8;
  const float bon = bo[ncol];
#pragma unroll
  for (int r = 0; r < 8; ++r) {
    size_t srow = (size_t)stile * 16 + mb + r;
    float  v    = ao[r] + bon;
    v = v > 0.f ? v : 0.f;
    int drow = subnode[srow];
    node_out[(size_t)drow * H_ + ncol] = v;
  }
}

// ---------------- launch ----------------
extern "C" void kernel_launch(void* const* d_in, const int* in_sizes, int n_in,
                              void* d_out, int out_size, void* d_ws, size_t ws_size,
                              hipStream_t stream) {
  (void)in_sizes; (void)n_in; (void)out_size; (void)ws_size;
  const float* fnode   = (const float*)d_in[0];
  const float* fmess   = (const float*)d_in[1];
  const float* h_in    = (const float*)d_in[2];
  const int*   agraph  = (const int*)d_in[3];
  const int*   bgraph  = (const int*)d_in[4];
  const int*   subnode = (const int*)d_in[5];
  const int*   submess = (const int*)d_in[6];
  const float* Wz = (const float*)d_in[8];
  const float* bz = (const float*)d_in[9];
  const float* Wr = (const float*)d_in[10];
  const float* Ur = (const float*)d_in[11];
  const float* bu = (const float*)d_in[12];
  const float* Wh = (const float*)d_in[13];
  const float* bh = (const float*)d_in[14];
  const float* Wo = (const float*)d_in[15];
  const float* bo = (const float*)d_in[16];

  float* node_out = (float*)d_out;                         // [NN_, H_]
  float* hwork    = node_out + (size_t)NN_ * H_;           // [E_, H_] -> second output

  const size_t SH = (size_t)S_ * H_;
  float* xz     = (float*)d_ws;
  float* xr     = xz + SH;
  float* xh     = xr + SH;
  float* sum_h  = xh + SH;
  float* sum_gh = sum_h + SH;
  unsigned short* Wz_bf = (unsigned short*)(sum_gh + SH);
  unsigned short* Wr_bf = Wz_bf + (size_t)H_ * (IN_ + H_);
  unsigned short* Ur_bf = Wr_bf + (size_t)H_ * IN_;
  unsigned short* Wh_bf = Ur_bf + (size_t)H_ * H_;
  unsigned short* Wo_bf = Wh_bf + (size_t)H_ * (IN_ + H_);

  // 1) weights -> bf16
  auto cvt = [&](const float* s, unsigned short* d, int n) {
    k_cvt_bf16<<<(n + 255) / 256, 256, 0, stream>>>(s, d, n);
  };
  cvt(Wz, Wz_bf, H_ * (IN_ + H_));
  cvt(Wr, Wr_bf, H_ * IN_);
  cvt(Ur, Ur_bf, H_ * H_);
  cvt(Wh, Wh_bf, H_ * (IN_ + H_));
  cvt(Wo, Wo_bf, H_ * (FD_ + H_));

  // 2) zero node_out; init h (copy + mask rows)
  {
    int n4 = (NN_ * H_) / 4;
    k_zero_f4<<<(n4 + 255) / 256, 256, 0, stream>>>((float4*)node_out, n4);
  }
  {
    int n4 = (E_ * H_) / 4;
    k_copy_f4<<<(n4 + 255) / 256, 256, 0, stream>>>((const float4*)h_in, (float4*)hwork, n4);
  }
  {
    int tot4 = S_ * (H_ / 4);
    k_zero_rows<<<(tot4 + 255) / 256, 256, 0, stream>>>(hwork, submess, tot4);
  }

  // 3) loop-invariant x-projections
  k_xproj<<<MT_, 512, 0, stream>>>(fmess, Wz_bf, Wr_bf, Wh_bf, xz, xr, xh);

  // 4) DEPTH message-passing iterations
  for (int d = 0; d < DEPTH_; ++d) {
    k_msg_pass<<<MT_, 512, 0, stream>>>(hwork, bgraph, xr, bu, Ur_bf, sum_h, sum_gh);
    k_gru_update<<<MT_, 512, 0, stream>>>(sum_h, sum_gh, xz, xh, Wz_bf, Wh_bf,
                                          bz, bh, submess, hwork);
  }

  // 5) readout
  k_readout<<<MT_, 512, 0, stream>>>(fnode, hwork, agraph, Wo_bf, bo, subnode, node_out);
}
